// TransformerEncode_16209206575751
// MI455X (gfx1250) — matrix operand order repaired and checked by reading, not compiled, not run on hardware
//
#include <hip/hip_runtime.h>

typedef unsigned short u16t;
typedef __attribute__((ext_vector_type(16))) __bf16 v16bf;
typedef __attribute__((ext_vector_type(8)))  float  v8f;

#define BB 2
#define SS 2048
#define HH 1024
#define NHH 16
#define DKK 64
#define MM (BB*SS)   /* 4096 */

// ---------------- helpers ----------------

__device__ inline u16t f2b(float f){                 // f32 -> bf16 RNE
  unsigned u = __float_as_uint(f);
  u += 0x7FFFu + ((u >> 16) & 1u);
  return (u16t)(u >> 16);
}

__device__ inline v8f vzero8(){
  v8f z = {0.f,0.f,0.f,0.f,0.f,0.f,0.f,0.f};
  return z;
}

__device__ inline v8f wmma_bf16(v16bf a, v16bf b, v8f c){
  // D = A(16x32 bf16) * B(32x16 bf16) + C(16x16 f32)
  return __builtin_amdgcn_wmma_f32_16x16x32_bf16(false, a, false, b, (short)0, c, false, false);
}

// wave-local LDS RAW fence (DS ops are in-order per wave; keep compiler honest)
__device__ inline void wave_lds_fence(){
  __builtin_amdgcn_wave_barrier();
  asm volatile("s_wait_dscnt 0" ::: "memory");
  __builtin_amdgcn_wave_barrier();
}

// CDNA5 async copy: 16B global -> LDS, tracked by ASYNCcnt (no VGPR round trip).
// GV mode: 64-bit global address in VGPR pair; VDST holds LDS byte offset.
__device__ inline void async_copy_b128(unsigned lds_addr, const void* gaddr){
  asm volatile("global_load_async_to_lds_b128 %0, %1, off"
               :: "v"(lds_addr), "v"((unsigned long long)(uintptr_t)gaddr)
               : "memory");
}

__device__ inline void wait_async0(){
  asm volatile("s_wait_asynccnt 0x0" ::: "memory");
}

// generic pointer to LDS: low 32 bits are the LDS byte offset
__device__ inline unsigned lds_off(const void* p){
  return (unsigned)(uintptr_t)p;
}

// A fragment: 16x32 bf16 tile, row-major source with `pitch` elements/row.
// Lane l (l<16): M=l,   K = {0..7, 16..23}
// Lane l (>=16): M=l-16,K = {8..15,24..31}
__device__ inline v16bf load_a_frag(const u16t* base, int pitch){
  int l   = threadIdx.x & 31;
  int row = l & 15;
  int kb  = (l < 16) ? 0 : 8;
  union { v16bf v; u16t u[16]; } r;
  const u16t* p0 = base + row*pitch + kb;
  const u16t* p1 = p0 + 16;
  #pragma unroll
  for (int i=0;i<8;i++){ r.u[i] = p0[i]; r.u[8+i] = p1[i]; }
  return r.v;
}

// B fragment: 32x16 bf16, source stored TRANSPOSED as [n][k] with `pitch` over k.
// Lane l: N = l&15, K = (l<16 ? 0 : 8) + {0..7, 16..23}
__device__ inline v16bf load_b_fragT(const u16t* baseT, int pitch){
  int l  = threadIdx.x & 31;
  int n  = l & 15;
  int kb = (l < 16) ? 0 : 8;
  union { v16bf v; u16t u[16]; } r;
  const u16t* p0 = baseT + n*pitch + kb;
  #pragma unroll
  for (int i=0;i<8;i++){ r.u[i] = p0[i]; r.u[8+i] = p0[16+i]; }
  return r.v;
}

// ---------------- elementwise kernels ----------------

__global__ void conv_kernel(const float* __restrict__ in, u16t* __restrict__ out, int n){
  int i = blockIdx.x*blockDim.x + threadIdx.x;
  if (i < n) out[i] = f2b(in[i]);
}

// out[n*K + k] = bf16(in[k*N + n])  : f32 -> bf16 with transpose (weights, one-time)
__global__ void convT_kernel(const float* __restrict__ in, u16t* __restrict__ out, int K, int N){
  int i = blockIdx.x*blockDim.x + threadIdx.x;
  if (i < K*N){
    int n = i / K, k = i - n*K;
    out[i] = f2b(in[(size_t)k*N + n]);
  }
}

__global__ void gelu_kernel(const float* __restrict__ in, u16t* __restrict__ out, int n){
  int i = blockIdx.x*blockDim.x + threadIdx.x;
  if (i < n){
    float x = in[i];
    float s = 1.0f / (1.0f + __expf(-1.702f * x));
    out[i] = f2b(x * s);
  }
}

// y = LN(a + res) * gamma + beta  (torch unbiased var, ddof=1); one row/block
__global__ __launch_bounds__(256) void ln_kernel(
    const float* __restrict__ a, const float* __restrict__ res,
    const float* __restrict__ gamma, const float* __restrict__ beta,
    float* __restrict__ outF, u16t* outB)
{
  int row = blockIdx.x;
  int t   = threadIdx.x;
  __shared__ float red[2][8];
  const float* pa = a   + (size_t)row*HH;
  const float* pr = res + (size_t)row*HH;
  float x[4], s = 0.f, ss = 0.f;
  #pragma unroll
  for (int i=0;i<4;i++){
    float v = pa[t + i*256] + pr[t + i*256];
    x[i] = v; s += v; ss += v*v;
  }
  #pragma unroll
  for (int off=1; off<32; off<<=1){
    s  += __shfl_xor(s,  off, 32);
    ss += __shfl_xor(ss, off, 32);
  }
  int wave = t >> 5, lane = t & 31;
  if (lane == 0){ red[0][wave] = s; red[1][wave] = ss; }
  __syncthreads();
  if (wave == 0){
    float s2  = (lane < 8) ? red[0][lane] : 0.f;
    float ss2 = (lane < 8) ? red[1][lane] : 0.f;
    #pragma unroll
    for (int off=1; off<8; off<<=1){
      s2  += __shfl_xor(s2,  off, 8);
      ss2 += __shfl_xor(ss2, off, 8);
    }
    if (lane == 0){ red[0][0] = s2; red[1][0] = ss2; }
  }
  __syncthreads();
  float sumv  = red[0][0];
  float sumsq = red[1][0];
  float mean = sumv * (1.0f/(float)HH);
  float var  = (sumsq - (float)HH*mean*mean) * (1.0f/((float)HH - 1.0f));
  float inv  = rsqrtf(var + 1e-12f);
  #pragma unroll
  for (int i=0;i<4;i++){
    int c = t + i*256;
    float y = gamma[c] * ((x[i]-mean)*inv) + beta[c];
    size_t idx = (size_t)row*HH + c;
    outF[idx] = y;
    if (outB) outB[idx] = f2b(y);
  }
}

// ---------------- WMMA GEMM:  C[M,N] = A[M,K]*W[K,N] + bias ----------------
// W supplied PRE-TRANSPOSED as WT[N][K]. WG 256 thr = 8 waves as 4(M)x2(N);
// WG tile 128x64; wave tile 32x32 (2x2 WMMA). Double-buffered async LDS staging.

template<bool OUTB>
__global__ __launch_bounds__(256) void gemm_bf16_kernel(
    const u16t* __restrict__ A, const u16t* __restrict__ WT,
    const float* __restrict__ bias,
    float* outF, u16t* outB, float scale, int M, int N, int K)
{
  __shared__ u16t As[2][128*32];   // A tile, row-major   (8KB each)
  __shared__ u16t Wt[2][64*32];    // WT tile [n][k]      (4KB each)
  int t = threadIdx.x;
  int wave = t >> 5, l = t & 31;
  int wm = wave >> 1, wn = wave & 1;
  int tileM = blockIdx.y * 128;
  int tileN = blockIdx.x * 64;
  v8f acc00 = vzero8(), acc01 = vzero8(), acc10 = vzero8(), acc11 = vzero8();

  // staging assignments
  int ar  = t >> 1;            // 0..127 A-tile row
  int ah  = (t & 1) * 16;      // 16-elem half of 32-wide k run
  int wr  = t >> 2;            // 0..63  WT-tile row (= n local)
  int wq  = (t & 3) * 8;       // 8-elem quarter of 32-wide k run

  const u16t* Arow = A  + (size_t)(tileM + ar)*K + ah;
  const u16t* Wrow = WT + (size_t)(tileN + wr)*K + wq;

  // prefetch tile 0 into buffer 0 (3 async b128 per thread)
  async_copy_b128(lds_off(&As[0][ar*32 + ah]),     Arow);
  async_copy_b128(lds_off(&As[0][ar*32 + ah + 8]), Arow + 8);
  async_copy_b128(lds_off(&Wt[0][wr*32 + wq]),     Wrow);

  int NT = K >> 5;
  for (int it = 0; it < NT; ++it){
    int cur = it & 1;
    wait_async0();            // my async stores for buf `cur` complete
    __syncthreads();          // everyone's complete; also everyone done reading buf cur^1
    if (it + 1 < NT){         // overlap next tile's fetch with this tile's WMMAs
      int k0 = (it + 1) << 5;
      async_copy_b128(lds_off(&As[cur^1][ar*32 + ah]),     Arow + k0);
      async_copy_b128(lds_off(&As[cur^1][ar*32 + ah + 8]), Arow + k0 + 8);
      async_copy_b128(lds_off(&Wt[cur^1][wr*32 + wq]),     Wrow + k0);
    }
    v16bf a0 = load_a_frag(&As[cur][(wm*32    )*32], 32);
    v16bf a1 = load_a_frag(&As[cur][(wm*32+16 )*32], 32);
    v16bf b0 = load_b_fragT(&Wt[cur][(wn*32   )*32], 32);
    v16bf b1 = load_b_fragT(&Wt[cur][(wn*32+16)*32], 32);
    acc00 = wmma_bf16(a0, b0, acc00);
    acc01 = wmma_bf16(a0, b1, acc01);
    acc10 = wmma_bf16(a1, b0, acc10);
    acc11 = wmma_bf16(a1, b1, acc11);
    __syncthreads();          // all reads of buf cur done before it is refilled
  }

  int hh = l >> 4, col = l & 15;
  #pragma unroll
  for (int i=0;i<2;i++){
    #pragma unroll
    for (int j=0;j<2;j++){
      v8f ac = (i==0) ? (j==0?acc00:acc01) : (j==0?acc10:acc11);
      int gcol = tileN + wn*32 + j*16 + col;
      float bv = bias[gcol];
      #pragma unroll
      for (int r=0;r<8;r++){
        int grow = tileM + wm*32 + i*16 + hh*8 + r;   // C layout: M = r + 8*(lane>=16)
        float v = ac[r] + bv;
        size_t idx = (size_t)grow*N + gcol;
        if (OUTB) outB[idx] = f2b(v*scale);
        else      outF[idx] = v;
      }
    }
  }
}

// ---------------- causal flash attention ----------------
// One wave per 16-row q tile of one (b,h). Q pre-scaled by 1/sqrt(DK).

__global__ __launch_bounds__(256) void attn_kernel(
    const u16t* __restrict__ Q, const u16t* __restrict__ Km,
    const u16t* __restrict__ Vm, float* __restrict__ O)
{
  __shared__ u16t Pl[8][16*32];   // P tile (16 q x 32 kv), per wave
  __shared__ u16t Vt[8][64*32];   // V tile transposed [d][kv], per wave
  int wave = threadIdx.x >> 5;
  int l    = threadIdx.x & 31;
  int g    = blockIdx.x * 8 + wave;       // 4096 q-tiles total
  int qt = g & 127;
  int h  = (g >> 7) & 15;
  int b  = g >> 11;
  int q0 = qt * 16;
  int hh = l >> 4, col = l & 15;

  const u16t* Qbase = Q + (size_t)(b*SS + q0)*HH + h*DKK;
  v16bf aq0 = load_a_frag(Qbase,      HH);   // d = 0..31
  v16bf aq1 = load_a_frag(Qbase + 32, HH);   // d = 32..63

  v8f acc[4];
  #pragma unroll
  for (int nt=0;nt<4;nt++) acc[nt] = vzero8();
  float mrow[8], lrow[8];
  #pragma unroll
  for (int r=0;r<8;r++){ mrow[r] = -1e30f; lrow[r] = 0.f; }

  u16t* P  = Pl[wave];
  u16t* Vw = Vt[wave];

  for (int kv0 = 0; kv0 < q0 + 16; kv0 += 32){
    // ---- scores: two 16x16 tiles (kv0..+15, kv0+16..+31) ----
    v8f s0 = vzero8(), s1 = vzero8();
    const u16t* Kb0 = Km + (size_t)(b*SS + kv0)*HH + h*DKK;
    const u16t* Kb1 = Kb0 + (size_t)16*HH;
    s0 = wmma_bf16(aq0, load_b_fragT(Kb0,      HH), s0);
    s0 = wmma_bf16(aq1, load_b_fragT(Kb0 + 32, HH), s0);
    s1 = wmma_bf16(aq0, load_b_fragT(Kb1,      HH), s1);
    s1 = wmma_bf16(aq1, load_b_fragT(Kb1 + 32, HH), s1);

    // ---- causal mask + online softmax ----
    float corr[8];
    #pragma unroll
    for (int r=0;r<8;r++){
      int qrow = q0 + hh*8 + r;
      int c0 = kv0 + col, c1 = c0 + 16;
      float x0 = (c0 <= qrow) ? (float)s0[r] : -1e30f;
      float x1 = (c1 <= qrow) ? (float)s1[r] : -1e30f;
      float mr = fmaxf(x0, x1);
      #pragma unroll
      for (int off=1; off<16; off<<=1) mr = fmaxf(mr, __shfl_xor(mr, off, 16));
      float mn = fmaxf(mrow[r], mr);
      corr[r] = __expf(mrow[r] - mn);
      mrow[r] = mn;
      float p0 = __expf(x0 - mn);
      float p1 = __expf(x1 - mn);
      float rs = p0 + p1;
      #pragma unroll
      for (int off=1; off<16; off<<=1) rs += __shfl_xor(rs, off, 16);
      lrow[r] = lrow[r]*corr[r] + rs;
      int rowm = hh*8 + r;
      P[rowm*32 + col]      = f2b(p0);
      P[rowm*32 + 16 + col] = f2b(p1);
      acc[0][r] *= corr[r]; acc[1][r] *= corr[r];
      acc[2][r] *= corr[r]; acc[3][r] *= corr[r];
    }

    // ---- stage V tile transposed: Vw[d][kv], kv row per lane ----
    {
      const u16t* Vrow = Vm + (size_t)(b*SS + kv0 + l)*HH + h*DKK;
      #pragma unroll
      for (int d0=0; d0<64; d0+=8){
        union { uint4 q; u16t u[8]; } tv;
        tv.q = *(const uint4*)(Vrow + d0);
        #pragma unroll
        for (int i=0;i<8;i++) Vw[(d0+i)*32 + l] = tv.u[i];
      }
    }
    wave_lds_fence();

    // ---- PV: acc[nt] += P(16x32) * V(32x16) ----
    v16bf ap = load_a_frag(P, 32);
    #pragma unroll
    for (int nt=0;nt<4;nt++)
      acc[nt] = wmma_bf16(ap, load_b_fragT(&Vw[nt*16*32], 32), acc[nt]);
    wave_lds_fence();
  }

  // ---- finalize & store ----
  #pragma unroll
  for (int nt=0;nt<4;nt++){
    #pragma unroll
    for (int r=0;r<8;r++){
      int grow = b*SS + q0 + hh*8 + r;
      O[(size_t)grow*HH + h*DKK + nt*16 + col] = acc[nt][r] / lrow[r];
    }
  }
}

// ---------------- driver ----------------

extern "C" void kernel_launch(void* const* d_in, const int* in_sizes, int n_in,
                              void* d_out, int out_size, void* d_ws, size_t ws_size,
                              hipStream_t stream)
{
  (void)in_sizes; (void)n_in; (void)out_size; (void)ws_size;
  const float* batch_x = (const float*)d_in[0];
  /* d_in[1] = mask_x: all-ones padding mask; causal handled in-kernel */
  const float* Wq  = (const float*)d_in[2];
  const float* bq  = (const float*)d_in[3];
  const float* Wk  = (const float*)d_in[4];
  const float* bk  = (const float*)d_in[5];
  const float* Wv  = (const float*)d_in[6];
  const float* bv  = (const float*)d_in[7];
  const float* Wo  = (const float*)d_in[8];
  const float* bo  = (const float*)d_in[9];
  const float* ln1w = (const float*)d_in[10];
  const float* ln1b = (const float*)d_in[11];
  const float* W1  = (const float*)d_in[12];
  const float* b1  = (const float*)d_in[13];
  const float* W2  = (const float*)d_in[14];
  const float* b2  = (const float*)d_in[15];
  const float* ln2w = (const float*)d_in[16];
  const float* ln2b = (const float*)d_in[17];
  float* out = (float*)d_out;

  char* ws = (char*)d_ws;
  u16t*  WB = (u16t*)ws;                         // 6 x 1M bf16 (transposed) = 12 MB
  u16t*  B0 = (u16t*)(ws + (size_t)(12u<<20));   // 8 MB each
  u16t*  B1 = (u16t*)(ws + (size_t)(20u<<20));
  u16t*  B2 = (u16t*)(ws + (size_t)(28u<<20));
  u16t*  B3 = (u16t*)(ws + (size_t)(36u<<20));
  float* F0 = (float*)(ws + (size_t)(44u<<20));  // 16 MB each
  float* F1 = (float*)(ws + (size_t)(60u<<20));

  const int HW = HH*HH;          // 1M elems per weight
  const int NX = MM*HH;          // 4M activation elems
  dim3 blk(256);
  dim3 gG(HH/64, MM/128);        // GEMM grid: 16 x 32

  // weights: f32 -> bf16 with transpose (WT[n][k]); activations: f32 -> bf16
  convT_kernel<<<(HW+255)/256, blk, 0, stream>>>(Wq, WB + 0*(size_t)HW, HH, HH);
  convT_kernel<<<(HW+255)/256, blk, 0, stream>>>(Wk, WB + 1*(size_t)HW, HH, HH);
  convT_kernel<<<(HW+255)/256, blk, 0, stream>>>(Wv, WB + 2*(size_t)HW, HH, HH);
  convT_kernel<<<(HW+255)/256, blk, 0, stream>>>(Wo, WB + 3*(size_t)HW, HH, HH);
  convT_kernel<<<(HW+255)/256, blk, 0, stream>>>(W1, WB + 4*(size_t)HW, HH, HH);
  convT_kernel<<<(HW+255)/256, blk, 0, stream>>>(W2, WB + 5*(size_t)HW, HH, HH);
  conv_kernel<<<(NX+255)/256, blk, 0, stream>>>(batch_x, B0, NX);

  // QKV projections (Q pre-scaled by 1/sqrt(DK) = 0.125)
  gemm_bf16_kernel<true ><<<gG, blk, 0, stream>>>(B0, WB + 0*(size_t)HW, bq, nullptr, B1, 0.125f, MM, HH, HH);
  gemm_bf16_kernel<true ><<<gG, blk, 0, stream>>>(B0, WB + 1*(size_t)HW, bk, nullptr, B2, 1.0f,   MM, HH, HH);
  gemm_bf16_kernel<true ><<<gG, blk, 0, stream>>>(B0, WB + 2*(size_t)HW, bv, nullptr, B3, 1.0f,   MM, HH, HH);

  // causal flash attention -> F0 (f32, [B,S,H]) ; 4096 q-tiles / 8 waves per WG
  attn_kernel<<<512, 256, 0, stream>>>(B1, B2, B3, F0);

  // output projection
  conv_kernel<<<(NX+255)/256, blk, 0, stream>>>(F0, B0, NX);
  gemm_bf16_kernel<false><<<gG, blk, 0, stream>>>(B0, WB + 3*(size_t)HW, bo, F1, nullptr, 1.0f, MM, HH, HH);

  // x1 = LN(attn + x)  -> F0 (f32) and B0 (bf16)
  ln_kernel<<<MM, 256, 0, stream>>>(F1, batch_x, ln1w, ln1b, F0, B0);

  // FFN
  gemm_bf16_kernel<false><<<gG, blk, 0, stream>>>(B0, WB + 4*(size_t)HW, b1, F1, nullptr, 1.0f, MM, HH, HH);
  gelu_kernel<<<(NX+255)/256, blk, 0, stream>>>(F1, B1, NX);
  gemm_bf16_kernel<false><<<gG, blk, 0, stream>>>(B1, WB + 5*(size_t)HW, b2, F1, nullptr, 1.0f, MM, HH, HH);

  // out = LN(ff + x1)
  ln_kernel<<<MM, 256, 0, stream>>>(F1, F0, ln2w, ln2b, out, nullptr);
}
